// DecoderLLM_14405320311563
// MI455X (gfx1250) — matrix operand order repaired and checked by reading, not compiled, not run on hardware
//
#include <hip/hip_runtime.h>
#include <cstdint>

typedef unsigned short u16;
typedef unsigned int   u32;

typedef __attribute__((ext_vector_type(16))) __bf16 bfvec16;
typedef __attribute__((ext_vector_type(8)))  float  v8f;
typedef __attribute__((ext_vector_type(8)))  u32    v8u;

constexpr int CB = 2, CS = 2048, CD = 1024, CH = 16, CDH = 64;
constexpr int CE = 8, CF = 4096, CDEPTH = 2;
constexpr int CT = CB * CS;                 // 4096 tokens
constexpr size_t MB = 1u << 20;

// workspace layout (bytes)
constexpr size_t OFF_XB    = 0;             //  8 MB  bf16 x
constexpr size_t OFF_R1    = 8*MB;          // 64 MB  q/k/v f32 (48MB) -> later hbuf bf16 (64MB)
constexpr size_t OFF_R2    = 72*MB;         // 32 MB  qb/kb/vb bf16 (24MB) -> later ybuf f32 (32MB)
constexpr size_t OFF_ATTN  = 104*MB;        // 16 MB  attn out f32
constexpr size_t OFF_Y1    = 120*MB;        // 16 MB  attn@Wo + x
constexpr size_t OFF_X1    = 136*MB;        // 16 MB  inter-layer ping buffer
constexpr size_t OFF_SMALL = 152*MB;        // routing metadata (~128KB)

// async global->LDS copy of 16 bytes (ASYNCcnt tracked), CDNA5 §15.18.3
#define ASYNC_LD_B128(ldsptr, gptr)                                            \
  do {                                                                         \
    unsigned __lds = (unsigned)(uintptr_t)(ldsptr);                            \
    unsigned long long __ga = (unsigned long long)(uintptr_t)(gptr);           \
    asm volatile("global_load_async_to_lds_b128 %0, %1, off"                   \
                 :: "v"(__lds), "v"(__ga) : "memory");                         \
  } while (0)
#define WAIT_ASYNC() asm volatile("s_wait_asynccnt 0x0" ::: "memory")

static __device__ __forceinline__ u16 f2bf(float f) {
  u32 u = __builtin_bit_cast(u32, f);
  u += 0x7FFFu + ((u >> 16) & 1u);          // RNE
  return (u16)(u >> 16);
}
static __device__ __forceinline__ float bf2f(u16 h) {
  return __builtin_bit_cast(float, (u32)h << 16);
}
static __device__ __forceinline__ v8f wmma_bf16(v8u a, v8u b, v8f c) {
  return __builtin_amdgcn_wmma_f32_16x16x32_bf16(
      false, __builtin_bit_cast(bfvec16, a),
      false, __builtin_bit_cast(bfvec16, b),
      (short)0, c, false, false);
}

// ---------------------------------------------------------------- cast f32->bf16
__global__ __launch_bounds__(256) void k_cast(const float* __restrict__ in,
                                              u16* __restrict__ out, int n) {
  int i = (blockIdx.x * 256 + threadIdx.x) * 4;
  if (i < n) {
    float4 f = *reinterpret_cast<const float4*>(in + i);
    out[i+0] = f2bf(f.x); out[i+1] = f2bf(f.y);
    out[i+2] = f2bf(f.z); out[i+3] = f2bf(f.w);
  }
}

// ---------------------------------------------------------------- WMMA GEMM
// C[128 x 64] tile per block (8 waves, each 16 rows x 64 cols). K step = 32.
// Double-buffered LDS; bf16 A staged with async global->LDS; f32 weights
// converted to bf16 while staging; next-next weight tile prefetched.
template<bool AF32, bool EXPERT, bool GATHER, bool GELU_, bool BIAS_, bool RESID_, bool OUTBF>
__global__ __launch_bounds__(256) void k_gemm(
    const u16* __restrict__ Abf, const float* __restrict__ Af32,
    const float* __restrict__ Bw, long long bstride,
    const float* __restrict__ bias, const float* __restrict__ resid,
    float* __restrict__ Cf, u16* __restrict__ Cb,
    const int* __restrict__ tok_idx, const int* __restrict__ eoff,
    const int* __restrict__ ecnt, int Mtot, int Nld, int Kd)
{
  __shared__ u16 Asl[2][128][56];   // K-pairs contiguous; 112B rows (16B aligned)
  __shared__ u16 Bst[2][64][40];    // [col][k], K-pairs contiguous
  const int tid = threadIdx.x;
  const int e = EXPERT ? blockIdx.z : 0;
  int Mv, rowbase;
  if constexpr (EXPERT) {
    int c = ecnt[e];
    int m0 = blockIdx.y * 128;
    if (m0 >= c) return;
    Mv = c - m0; if (Mv > 128) Mv = 128;
    rowbase = eoff[e] + m0;
  } else {
    rowbase = blockIdx.y * 128;
    Mv = Mtot - rowbase; if (Mv > 128) Mv = 128;
  }
  const float* Bp = Bw + (EXPERT ? (long long)e * bstride : 0);
  const int col0 = blockIdx.x * 64;

  // A-tile loader role: row = tid>>1, 16 elements starting at (tid&1)*16
  const int lrow = tid >> 1;
  const int lr_c = (lrow < Mv) ? lrow : (Mv - 1);
  int tokA;
  if constexpr (GATHER) tokA = tok_idx[rowbase + lr_c];
  else                  tokA = rowbase + lr_c;
  const int ksegA = (tid & 1) * 16;
  // B-tile loader role: k = tid>>3, 8 cols starting at (tid&7)*8
  const int bk  = tid >> 3;
  const int bc0 = (tid & 7) * 8;

  const int lane = tid & 31, wv = tid >> 5;
  const int wr = wv * 16, l = lane & 15, hi = lane >> 4;

  const int nK = Kd >> 5;

  auto stageA = [&](int kc, int buf) {
    int kg = kc * 32 + ksegA;
    if constexpr (AF32) {
      const float4* p = reinterpret_cast<const float4*>(Af32 + (size_t)tokA * Kd + kg);
      float4 f0 = p[0], f1 = p[1], f2 = p[2], f3 = p[3];
      u32* d = reinterpret_cast<u32*>(&Asl[buf][lrow][ksegA]);
      d[0] = (u32)f2bf(f0.x) | ((u32)f2bf(f0.y) << 16);
      d[1] = (u32)f2bf(f0.z) | ((u32)f2bf(f0.w) << 16);
      d[2] = (u32)f2bf(f1.x) | ((u32)f2bf(f1.y) << 16);
      d[3] = (u32)f2bf(f1.z) | ((u32)f2bf(f1.w) << 16);
      d[4] = (u32)f2bf(f2.x) | ((u32)f2bf(f2.y) << 16);
      d[5] = (u32)f2bf(f2.z) | ((u32)f2bf(f2.w) << 16);
      d[6] = (u32)f2bf(f3.x) | ((u32)f2bf(f3.y) << 16);
      d[7] = (u32)f2bf(f3.z) | ((u32)f2bf(f3.w) << 16);
    } else {
      ASYNC_LD_B128(&Asl[buf][lrow][ksegA], Abf + (size_t)tokA * Kd + kg);
      ASYNC_LD_B128(&Asl[buf][lrow][ksegA + 8], Abf + (size_t)tokA * Kd + kg + 8);
    }
  };
  auto stageB = [&](int kc, int buf) {
    const float4* p = reinterpret_cast<const float4*>(Bp + (size_t)(kc*32 + bk) * Nld + col0 + bc0);
    float4 f0 = p[0], f1 = p[1];
    Bst[buf][bc0+0][bk] = f2bf(f0.x); Bst[buf][bc0+1][bk] = f2bf(f0.y);
    Bst[buf][bc0+2][bk] = f2bf(f0.z); Bst[buf][bc0+3][bk] = f2bf(f0.w);
    Bst[buf][bc0+4][bk] = f2bf(f1.x); Bst[buf][bc0+5][bk] = f2bf(f1.y);
    Bst[buf][bc0+6][bk] = f2bf(f1.z); Bst[buf][bc0+7][bk] = f2bf(f1.w);
  };

  v8f zz = {0.f,0.f,0.f,0.f,0.f,0.f,0.f,0.f};
  v8f acc[4]; acc[0] = zz; acc[1] = zz; acc[2] = zz; acc[3] = zz;

  stageA(0, 0);
  stageB(0, 0);
  WAIT_ASYNC();
  __syncthreads();

  for (int kc = 0; kc < nK; ++kc) {
    const int buf = kc & 1;
    if (kc + 1 < nK) {
      stageA(kc + 1, buf ^ 1);
      stageB(kc + 1, buf ^ 1);
      if (kc + 2 < nK)   // prefetch weight stream 2 chunks ahead
        __builtin_prefetch(Bp + (size_t)((kc+2)*32 + bk) * Nld + col0 + bc0, 0, 1);
    }
    v8u av;
    #pragma unroll
    for (int j = 0; j < 8; ++j) {
      int kp = ((j < 4) ? 2*j : 2*j + 8) + 8*hi;
      av[j] = *reinterpret_cast<const u32*>(&Asl[buf][wr + l][kp]);
    }
    #pragma unroll
    for (int nt = 0; nt < 4; ++nt) {
      v8u bv;
      #pragma unroll
      for (int j = 0; j < 8; ++j)
        bv[j] = *reinterpret_cast<const u32*>(&Bst[buf][nt*16 + l][((j < 4) ? 2*j : 2*j + 8) + 8*hi]);
      acc[nt] = wmma_bf16(av, bv, acc[nt]);
    }
    WAIT_ASYNC();
    __syncthreads();
  }
  // epilogue
  #pragma unroll
  for (int nt = 0; nt < 4; ++nt) {
    #pragma unroll
    for (int r = 0; r < 8; ++r) {
      int rl = wr + r + 8*hi;
      if (rl >= Mv) continue;
      size_t crow = (size_t)(rowbase + rl);
      int gcol = col0 + nt*16 + l;
      float v = acc[nt][r];
      if constexpr (BIAS_)  v += bias[(size_t)e * Nld + gcol];
      if constexpr (GELU_) {
        float v3 = v * v * v;
        v = 0.5f * v * (1.f + tanhf(0.7978845608f * (v + 0.044715f * v3)));
      }
      if constexpr (RESID_) v += resid[crow * Nld + gcol];
      if constexpr (OUTBF)  Cb[crow * Nld + gcol] = f2bf(v);
      else                  Cf[crow * Nld + gcol] = v;
    }
  }
}

// ------------------------------------------------- qk-norm / cast to [B][H][S][DH]
__global__ __launch_bounds__(256) void k_qknorm(const float* __restrict__ src,
                                                const float* __restrict__ scale,
                                                u16* __restrict__ dst,
                                                float mul, int do_norm) {
  int idx = blockIdx.x * 256 + threadIdx.x;       // CT*CH threads
  int t = idx >> 4, h = idx & 15;
  int b = t >> 11, s = t & 2047;
  const float* p = src + (size_t)t * CD + h * CDH;
  float4 buf[16];
  float ss = 0.f;
  #pragma unroll
  for (int i = 0; i < 16; ++i) {
    buf[i] = *reinterpret_cast<const float4*>(p + i*4);
    ss += buf[i].x*buf[i].x + buf[i].y*buf[i].y + buf[i].z*buf[i].z + buf[i].w*buf[i].w;
  }
  float r = do_norm ? rsqrtf(ss + 1e-6f) : 1.f;
  u16* q = dst + ((size_t)(b * CH + h) * CS + s) * CDH;
  #pragma unroll
  for (int i = 0; i < 16; ++i) {
    float4 sc = scale ? *reinterpret_cast<const float4*>(scale + i*4)
                      : make_float4(1.f,1.f,1.f,1.f);
    q[i*4+0] = f2bf(buf[i].x * r * sc.x * mul);
    q[i*4+1] = f2bf(buf[i].y * r * sc.y * mul);
    q[i*4+2] = f2bf(buf[i].z * r * sc.z * mul);
    q[i*4+3] = f2bf(buf[i].w * r * sc.w * mul);
  }
}

// ---------------------------------------------------------------- flash attention
// grid(S/64, B*H), block 128 (4 waves). Each wave: 16 q-rows. 32-key chunks.
__global__ __launch_bounds__(128) void k_flash(const u16* __restrict__ Qb,
                                               const u16* __restrict__ Kb,
                                               const u16* __restrict__ Vb,
                                               float* __restrict__ O) {
  __shared__ u16 Qs[64][72];
  __shared__ u16 Ks[32][72];
  __shared__ u16 Vst[64][40];
  __shared__ u16 Ps[4][16][40];
  const int tid = threadIdx.x;
  const int lane = tid & 31, wv = tid >> 5;
  const int l = lane & 15, hi = lane >> 4;
  const int bh = blockIdx.y, b = bh >> 4, h = bh & 15;
  const int q0 = blockIdx.x * 64;
  const size_t base = (size_t)bh * CS * CDH;

  { // stage Q tile (async global->LDS)
    int row = tid >> 1, seg = (tid & 1) * 32;
    const u16* src = Qb + base + (size_t)(q0 + row) * 64 + seg;
    u16* dst = &Qs[row][seg];
    ASYNC_LD_B128(dst,      src);
    ASYNC_LD_B128(dst + 8,  src + 8);
    ASYNC_LD_B128(dst + 16, src + 16);
    ASYNC_LD_B128(dst + 24, src + 24);
  }
  WAIT_ASYNC();
  __syncthreads();
  const int wr = wv * 16;
  v8u aq[2];
  #pragma unroll
  for (int dc = 0; dc < 2; ++dc)
    #pragma unroll
    for (int j = 0; j < 8; ++j) {
      int kp = dc*32 + ((j < 4) ? 2*j : 2*j + 8) + 8*hi;
      aq[dc][j] = *reinterpret_cast<const u32*>(&Qs[wr + l][kp]);
    }

  v8f zz = {0.f,0.f,0.f,0.f,0.f,0.f,0.f,0.f};
  v8f o[4]; o[0]=zz; o[1]=zz; o[2]=zz; o[3]=zz;
  float mrow[8], lrow[8];
  #pragma unroll
  for (int r = 0; r < 8; ++r) { mrow[r] = -3.0e38f; lrow[r] = 0.f; }

  const int nkb = (q0 + 64) >> 5;
  for (int kb = 0; kb < nkb; ++kb) {
    const int k0 = kb << 5;
    __syncthreads();
    { // stage K [key][d] (async)
      int key = tid >> 2, seg = (tid & 3) * 16;
      const u16* src = Kb + base + (size_t)(k0 + key) * 64 + seg;
      u16* dst = &Ks[key][seg];
      ASYNC_LD_B128(dst,     src);
      ASYNC_LD_B128(dst + 8, src + 8);
    }
    { // stage V transposed: Vst[d][key]
      int key = tid >> 2, seg = (tid & 3) * 16;
      const uint4* src = reinterpret_cast<const uint4*>(Vb + base + (size_t)(k0 + key) * 64 + seg);
      uint4 a0 = src[0], a1 = src[1];
      u32 wv8[8] = {a0.x,a0.y,a0.z,a0.w,a1.x,a1.y,a1.z,a1.w};
      #pragma unroll
      for (int i = 0; i < 8; ++i) {
        Vst[seg + 2*i    ][key] = (u16)(wv8[i] & 0xffffu);
        Vst[seg + 2*i + 1][key] = (u16)(wv8[i] >> 16);
      }
    }
    WAIT_ASYNC();
    __syncthreads();
    v8f st[2];
    #pragma unroll
    for (int t = 0; t < 2; ++t) {
      st[t] = zz;
      #pragma unroll
      for (int dc = 0; dc < 2; ++dc) {
        v8u bk;
        #pragma unroll
        for (int j = 0; j < 8; ++j)
          bk[j] = *reinterpret_cast<const u32*>(&Ks[t*16 + l][dc*32 + ((j < 4) ? 2*j : 2*j + 8) + 8*hi]);
        st[t] = wmma_bf16(aq[dc], bk, st[t]);
      }
    }
    #pragma unroll
    for (int r = 0; r < 8; ++r) {
      int rowg = q0 + wr + r + 8*hi;
      #pragma unroll
      for (int t = 0; t < 2; ++t) {
        int keyg = k0 + t*16 + l;
        if (keyg > rowg) st[t][r] = -3.0e38f;
      }
      float mx = fmaxf(st[0][r], st[1][r]);
      mx = fmaxf(mx, __shfl_xor(mx, 1, 32));
      mx = fmaxf(mx, __shfl_xor(mx, 2, 32));
      mx = fmaxf(mx, __shfl_xor(mx, 4, 32));
      mx = fmaxf(mx, __shfl_xor(mx, 8, 32));
      float mnew = fmaxf(mrow[r], mx);
      float sc = __expf(mrow[r] - mnew);
      float p0 = __expf(st[0][r] - mnew);
      float p1 = __expf(st[1][r] - mnew);
      float ps = p0 + p1;
      ps += __shfl_xor(ps, 1, 32);
      ps += __shfl_xor(ps, 2, 32);
      ps += __shfl_xor(ps, 4, 32);
      ps += __shfl_xor(ps, 8, 32);
      lrow[r] = lrow[r] * sc + ps;
      mrow[r] = mnew;
      #pragma unroll
      for (int nt = 0; nt < 4; ++nt) o[nt][r] *= sc;
      Ps[wv][r + 8*hi][l     ] = f2bf(p0);
      Ps[wv][r + 8*hi][16 + l] = f2bf(p1);
    }
    __syncthreads();
    v8u pa;
    #pragma unroll
    for (int j = 0; j < 8; ++j) {
      int kp = ((j < 4) ? 2*j : 2*j + 8) + 8*hi;
      pa[j] = *reinterpret_cast<const u32*>(&Ps[wv][l][kp]);
    }
    #pragma unroll
    for (int nt = 0; nt < 4; ++nt) {
      v8u bv;
      #pragma unroll
      for (int j = 0; j < 8; ++j)
        bv[j] = *reinterpret_cast<const u32*>(&Vst[nt*16 + l][((j < 4) ? 2*j : 2*j + 8) + 8*hi]);
      o[nt] = wmma_bf16(pa, bv, o[nt]);
    }
  }
  #pragma unroll
  for (int r = 0; r < 8; ++r) {
    float inv = 1.f / lrow[r];
    int rowg = q0 + wr + r + 8*hi;
    size_t trow = ((size_t)b * CS + rowg) * CD + h * CDH;
    #pragma unroll
    for (int nt = 0; nt < 4; ++nt)
      O[trow + nt*16 + l] = o[nt][r] * inv;
  }
}

// ---------------------------------------------------------------- MoE routing
__global__ void k_zero(int* cnt) { if (threadIdx.x < 8) cnt[threadIdx.x] = 0; }

__global__ __launch_bounds__(256) void k_gate(const u16* __restrict__ xb,
                                              const float* __restrict__ Wg,
                                              int* __restrict__ cnt,
                                              float* __restrict__ wout,
                                              int* __restrict__ eid) {
  int t = blockIdx.x * 256 + threadIdx.x;
  if (t >= CT) return;
  float acc[CE] = {0,0,0,0,0,0,0,0};
  const u16* xp = xb + (size_t)t * CD;
  for (int d = 0; d < CD; ++d) {
    float xv = bf2f(xp[d]);
    const float* wr = Wg + (size_t)d * CE;
    #pragma unroll
    for (int e = 0; e < CE; ++e) acc[e] += xv * wr[e];
  }
  int i0 = 0; float v0 = acc[0];
  #pragma unroll
  for (int e = 1; e < CE; ++e) if (acc[e] > v0) { v0 = acc[e]; i0 = e; }
  int i1 = -1; float v1 = -3.0e38f;
  #pragma unroll
  for (int e = 0; e < CE; ++e) if (e != i0 && acc[e] > v1) { v1 = acc[e]; i1 = e; }
  float w0 = 1.f / (1.f + __expf(v1 - v0));
  float w1 = 1.f - w0;
  eid[t*2] = i0; eid[t*2+1] = i1;
  wout[t*2] = w0; wout[t*2+1] = w1;
  atomicAdd(&cnt[i0], 1);
  atomicAdd(&cnt[i1], 1);
}

__global__ void k_offsets(const int* cnt, int* off, int* cur) {
  if (threadIdx.x == 0) { int a = 0; for (int e = 0; e < CE; ++e) { off[e] = a; a += cnt[e]; } }
  if (threadIdx.x < 8) cur[threadIdx.x] = 0;
}

__global__ __launch_bounds__(256) void k_slots(const int* __restrict__ eid,
                                               const int* __restrict__ off,
                                               int* __restrict__ cur,
                                               int* __restrict__ tok_of,
                                               int* __restrict__ slot_of) {
  int t = blockIdx.x * 256 + threadIdx.x;
  if (t >= CT) return;
  #pragma unroll
  for (int k = 0; k < 2; ++k) {
    int e = eid[t*2+k];
    int pos = atomicAdd(&cur[e], 1);
    int s = off[e] + pos;
    tok_of[s] = t;
    slot_of[t*2+k] = s;
  }
}

__global__ __launch_bounds__(256) void k_combine(const float* __restrict__ y1,
                                                 const float* __restrict__ ybuf,
                                                 const float* __restrict__ wk,
                                                 const int* __restrict__ slot_of,
                                                 float* __restrict__ out) {
  int i = blockIdx.x * 256 + threadIdx.x;      // over CT*CD/4
  int c4 = i * 4;
  int t = c4 >> 10, col = c4 & 1023;
  float w0 = wk[t*2], w1 = wk[t*2+1];
  int s0 = slot_of[t*2], s1 = slot_of[t*2+1];
  float4 a = *reinterpret_cast<const float4*>(y1 + (size_t)t * CD + col);
  float4 p = *reinterpret_cast<const float4*>(ybuf + (size_t)s0 * CD + col);
  float4 q = *reinterpret_cast<const float4*>(ybuf + (size_t)s1 * CD + col);
  float4 o;
  o.x = a.x + w0*p.x + w1*q.x; o.y = a.y + w0*p.y + w1*q.y;
  o.z = a.z + w0*p.z + w1*q.z; o.w = a.w + w0*p.w + w1*q.w;
  *reinterpret_cast<float4*>(out + (size_t)t * CD + col) = o;
}

// ---------------------------------------------------------------- launcher
extern "C" void kernel_launch(void* const* d_in, const int* in_sizes, int n_in,
                              void* d_out, int out_size, void* d_ws, size_t ws_size,
                              hipStream_t stream) {
  (void)in_sizes; (void)n_in; (void)out_size; (void)ws_size;
  const float* x   = (const float*)d_in[0];
  const float* Wq  = (const float*)d_in[1];
  const float* Wk  = (const float*)d_in[2];
  const float* Wv  = (const float*)d_in[3];
  const float* Wo  = (const float*)d_in[4];
  const float* qsc = (const float*)d_in[5];
  const float* ksc = (const float*)d_in[6];
  const float* Wg  = (const float*)d_in[7];
  const float* W1  = (const float*)d_in[8];
  const float* b1  = (const float*)d_in[9];
  const float* W2  = (const float*)d_in[10];
  const float* b2  = (const float*)d_in[11];

  char* w = (char*)d_ws;
  u16*   xb   = (u16*)  (w + OFF_XB);
  float* qf   = (float*)(w + OFF_R1);
  float* kf   = (float*)(w + OFF_R1 + 16*MB);
  float* vf   = (float*)(w + OFF_R1 + 32*MB);
  u16*   hbuf = (u16*)  (w + OFF_R1);
  u16*   qbh  = (u16*)  (w + OFF_R2);
  u16*   kbh  = (u16*)  (w + OFF_R2 + 8*MB);
  u16*   vbh  = (u16*)  (w + OFF_R2 + 16*MB);
  float* ybuf = (float*)(w + OFF_R2);
  float* attn = (float*)(w + OFF_ATTN);
  float* y1   = (float*)(w + OFF_Y1);
  float* x1   = (float*)(w + OFF_X1);
  int*   cnt  = (int*)  (w + OFF_SMALL);
  int*   cur  = cnt + 8;
  int*   off8 = cnt + 16;
  int*   eid  = cnt + 64;
  int*   slot_of = eid + 2*CT;
  int*   tok_of  = slot_of + 2*CT;
  float* wk      = (float*)(tok_of + 2*CT);

  const dim3 gp(CD/64, CT/128);
  for (int layer = 0; layer < CDEPTH; ++layer) {
    const float* xin = (layer == 0) ? x : x1;
    float* xout = (layer == CDEPTH-1) ? (float*)d_out : x1;
    const float* Wq_l = Wq + (size_t)layer * CD * CD;
    const float* Wk_l = Wk + (size_t)layer * CD * CD;
    const float* Wv_l = Wv + (size_t)layer * CD * CD;
    const float* Wo_l = Wo + (size_t)layer * CD * CD;
    const float* qs_l = qsc + (size_t)layer * CDH;
    const float* ks_l = ksc + (size_t)layer * CDH;
    const float* Wg_l = Wg + (size_t)layer * CD * CE;
    const float* W1_l = W1 + (size_t)layer * CE * CD * CF;
    const float* b1_l = b1 + (size_t)layer * CE * CF;
    const float* W2_l = W2 + (size_t)layer * CE * CF * CD;
    const float* b2_l = b2 + (size_t)layer * CE * CD;

    k_cast<<<dim3(CT*CD/1024), 256, 0, stream>>>(xin, xb, CT*CD);
    // QKV projections (bf16 WMMA, fp32 out)
    k_gemm<false,false,false,false,false,false,false><<<gp,256,0,stream>>>(
        xb, nullptr, Wq_l, 0, nullptr, nullptr, qf, nullptr, nullptr, nullptr, nullptr, CT, CD, CD);
    k_gemm<false,false,false,false,false,false,false><<<gp,256,0,stream>>>(
        xb, nullptr, Wk_l, 0, nullptr, nullptr, kf, nullptr, nullptr, nullptr, nullptr, CT, CD, CD);
    k_gemm<false,false,false,false,false,false,false><<<gp,256,0,stream>>>(
        xb, nullptr, Wv_l, 0, nullptr, nullptr, vf, nullptr, nullptr, nullptr, nullptr, CT, CD, CD);
    // qk-norm (QK_SCALE folded into q), layout -> [B][H][S][DH] bf16
    k_qknorm<<<dim3(CT*CH/256),256,0,stream>>>(qf, qs_l, qbh, 10.f, 1);
    k_qknorm<<<dim3(CT*CH/256),256,0,stream>>>(kf, ks_l, kbh, 1.f, 1);
    k_qknorm<<<dim3(CT*CH/256),256,0,stream>>>(vf, nullptr, vbh, 1.f, 0);
    // flash attention
    k_flash<<<dim3(CS/64, CB*CH), 128, 0, stream>>>(qbh, kbh, vbh, attn);
    // Wo projection + residual (A is fp32 attn output)
    k_gemm<true,false,false,false,false,true,false><<<gp,256,0,stream>>>(
        nullptr, attn, Wo_l, 0, nullptr, xin, y1, nullptr, nullptr, nullptr, nullptr, CT, CD, CD);
    // MoE routing
    k_zero<<<1,32,0,stream>>>(cnt);
    k_gate<<<dim3(CT/256),256,0,stream>>>(xb, Wg_l, cnt, wk, eid);
    k_offsets<<<1,32,0,stream>>>(cnt, off8, cur);
    k_slots<<<dim3(CT/256),256,0,stream>>>(eid, off8, cur, tok_of, slot_of);
    // FFN1: gathered rows, +b1, GELU, bf16 out
    k_gemm<false,true,true,true,true,false,true><<<dim3(CF/64, CT/128, CE),256,0,stream>>>(
        xb, nullptr, W1_l, (long long)CD*CF, b1_l, nullptr, nullptr, hbuf,
        tok_of, off8, cnt, CT, CF, CD);
    // FFN2: compact slots, +b2, fp32 out
    k_gemm<false,true,false,false,true,false,false><<<dim3(CD/64, CT/128, CE),256,0,stream>>>(
        hbuf, nullptr, W2_l, (long long)CF*CD, b2_l, nullptr, ybuf, nullptr,
        nullptr, off8, cnt, CT, CD, CF);
    // out = (attn@Wo + x) + sum_k w_k * expert_k
    k_combine<<<dim3(CT*CD/1024),256,0,stream>>>(y1, ybuf, wk, slot_of, xout);
  }
}
